// WS_LSTM_85177791414669
// MI455X (gfx1250) — matrix operand order repaired and checked by reading, not compiled, not run on hardware
//
#include <hip/hip_runtime.h>

typedef __attribute__((ext_vector_type(16))) __bf16 v16bf;
typedef __attribute__((ext_vector_type(8)))  __bf16 v8bf;
typedef __attribute__((ext_vector_type(8)))  float  v8f;
typedef __attribute__((ext_vector_type(4)))  float  v4f;

#define HDIM   1024
#define EDIM   1024
#define FOURH  4096
#define BSZ    1024
#define TSTEPS 64
#define LDA    40      // halves per LDS tile row (32 + 8 pad -> conflict-free)

// ---------------------------------------------------------------------------
// Prep kernels
// ---------------------------------------------------------------------------
__global__ __launch_bounds__(256) void f32_to_bf16_k(const float* __restrict__ src,
                                                     __bf16* __restrict__ dst, int n) {
    int i = blockIdx.x * 256 + threadIdx.x;
    if (i < n) dst[i] = (__bf16)src[i];
}

__global__ __launch_bounds__(256) void bias_sum_k(const float* __restrict__ bih,
                                                  const float* __restrict__ bhh,
                                                  float* __restrict__ bias, int n) {
    int i = blockIdx.x * 256 + threadIdx.x;
    if (i < n) bias[i] = bih[i] + bhh[i];
}

// ---------------------------------------------------------------------------
// Per-wave compute: 4(M) x 2(N) WMMA tiles from staged LDS tiles.
// A frag: lane l<16 -> row M=l, K=kh*8.. ; B frag: lane -> col N, same K as A.
// ---------------------------------------------------------------------------
__device__ __forceinline__
void compute_tiles(const __bf16* __restrict__ As, const __bf16* __restrict__ Bs,
                   v8f acc[4][2], int waveM, int waveN, int l15, int lhi)
{
    v16bf bfrag[2];
    #pragma unroll
    for (int nt = 0; nt < 2; ++nt) {
        const __bf16* p = &Bs[(waveN * 32 + nt * 16 + l15) * LDA];
        ((v8bf*)&bfrag[nt])[0] = *(const v8bf*)(p + lhi * 8);
        ((v8bf*)&bfrag[nt])[1] = *(const v8bf*)(p + 16 + lhi * 8);
    }
    #pragma unroll
    for (int mt = 0; mt < 4; ++mt) {
        const __bf16* p = &As[(waveM * 64 + mt * 16 + l15) * LDA];
        v16bf afrag;
        ((v8bf*)&afrag)[0] = *(const v8bf*)(p + lhi * 8);        // K = kh..kh+7
        ((v8bf*)&afrag)[1] = *(const v8bf*)(p + 16 + lhi * 8);   // K = 16+kh..
        #pragma unroll
        for (int nt = 0; nt < 2; ++nt) {
            acc[mt][nt] = __builtin_amdgcn_wmma_f32_16x16x32_bf16(
                false, afrag, false, bfrag[nt],
                (short)0, acc[mt][nt], false, false);
        }
    }
}

// ---------------------------------------------------------------------------
// WMMA GEMM: gates[B,4H] = [emb(x_t) | h] @ [W_ih | W_hh]^T + (b_ih + b_hh)
// Block: 256 threads (8 waves). Block tile 128(M) x 128(N). K-step 32.
// Double-buffered LDS, one barrier per K-step. Grid: (32, 8).
// ---------------------------------------------------------------------------
__global__ __launch_bounds__(256)
void gemm_gates_k(const int* __restrict__ x, const float* __restrict__ emb,
                  const __bf16* __restrict__ Wih, const __bf16* __restrict__ Whh,
                  const float* __restrict__ bias, const __bf16* __restrict__ hprev,
                  float* __restrict__ gates, int t)
{
    __shared__ __bf16 As[2][128 * LDA];
    __shared__ __bf16 Bs[2][128 * LDA];

    const int tid   = threadIdx.x;
    const int lane  = tid & 31;
    const int wave  = tid >> 5;
    const int waveM = wave & 1;                // 2 waves along M (64 rows each)
    const int waveN = wave >> 1;               // 4 waves along N (32 cols each)
    const int m0    = blockIdx.y * 128;
    const int n0    = blockIdx.x * 128;
    const int l15   = lane & 15;
    const int lhi   = lane >> 4;

    // accumulators initialized with bias (broadcast per output column)
    v8f acc[4][2];
    #pragma unroll
    for (int nt = 0; nt < 2; ++nt) {
        float bv = bias[n0 + waveN * 32 + nt * 16 + l15];
        v8f binit;
        #pragma unroll
        for (int r = 0; r < 8; ++r) binit[r] = bv;
        #pragma unroll
        for (int mt = 0; mt < 4; ++mt) acc[mt][nt] = binit;
    }

    // cooperative staging: 2 threads per tile row, 16 halves each (loop-invariant)
    const int srow = tid >> 1;                 // 0..127
    const int skh  = (tid & 1) * 16;           // 0 or 16 halves
    const int tok  = x[(m0 + srow) * TSTEPS + t];
    const float*  embrow = emb   + (size_t)tok * EDIM        + skh;
    const __bf16* hrow   = hprev + (size_t)(m0 + srow) * HDIM + skh;
    const __bf16* wihrow = Wih   + (size_t)(n0 + srow) * EDIM + skh;
    const __bf16* whhrow = Whh   + (size_t)(n0 + srow) * HDIM + skh;
    __bf16* const adst0  = &As[0][srow * LDA + skh];
    __bf16* const adst1  = &As[1][srow * LDA + skh];
    __bf16* const bdst0  = &Bs[0][srow * LDA + skh];
    __bf16* const bdst1  = &Bs[1][srow * LDA + skh];

    auto stageE = [&](int buf, int k0) {       // embedding phase: gather + cvt
        v4f f[4];
        const v4f* s4 = (const v4f*)(embrow + k0);
        #pragma unroll
        for (int q = 0; q < 4; ++q) f[q] = s4[q];
        // pack converts into two aligned 16B vectors -> two ds_store_b128
        v8bf lo, hi;
        #pragma unroll
        for (int i = 0; i < 4; ++i) { lo[i]     = (__bf16)f[0][i];
                                      lo[i + 4] = (__bf16)f[1][i];
                                      hi[i]     = (__bf16)f[2][i];
                                      hi[i + 4] = (__bf16)f[3][i]; }
        __bf16* ad = buf ? adst1 : adst0;
        *(v8bf*)(ad)     = lo;
        *(v8bf*)(ad + 8) = hi;
        __bf16* bd = buf ? bdst1 : bdst0;
        *(v8bf*)(bd)     = *(const v8bf*)(wihrow + k0);
        *(v8bf*)(bd + 8) = *(const v8bf*)(wihrow + k0 + 8);
    };
    auto stageH = [&](int buf, int k0) {       // hidden phase: bf16 copy
        __bf16* ad = buf ? adst1 : adst0;
        *(v8bf*)(ad)     = *(const v8bf*)(hrow + k0);
        *(v8bf*)(ad + 8) = *(const v8bf*)(hrow + k0 + 8);
        __bf16* bd = buf ? bdst1 : bdst0;
        *(v8bf*)(bd)     = *(const v8bf*)(whhrow + k0);
        *(v8bf*)(bd + 8) = *(const v8bf*)(whhrow + k0 + 8);
    };

    constexpr int NKE = EDIM / 32, NKH = HDIM / 32;
    int buf = 0;
    stageE(0, 0);
    // ---- phase 1: K over embedding columns
    for (int kt = 0; kt < NKE; ++kt) {
        __syncthreads();
        if (kt + 1 < NKE) stageE(buf ^ 1, (kt + 1) * 32);
        else              stageH(buf ^ 1, 0);
        compute_tiles(As[buf], Bs[buf], acc, waveM, waveN, l15, lhi);
        buf ^= 1;
    }
    // ---- phase 2: K over hidden-state columns
    for (int kt = 0; kt < NKH; ++kt) {
        __syncthreads();
        if (kt + 1 < NKH) stageH(buf ^ 1, (kt + 1) * 32);
        compute_tiles(As[buf], Bs[buf], acc, waveM, waveN, l15, lhi);
        buf ^= 1;
    }

    // ---- epilogue: D layout — VGPR r holds row (r + 8*lhi), lane holds col l15
    #pragma unroll
    for (int mt = 0; mt < 4; ++mt) {
        #pragma unroll
        for (int nt = 0; nt < 2; ++nt) {
            int col     = n0 + waveN * 32 + nt * 16 + l15;
            int rowbase = m0 + waveM * 64 + mt * 16 + lhi * 8;
            #pragma unroll
            for (int r = 0; r < 8; ++r)
                gates[(size_t)(rowbase + r) * FOURH + col] = acc[mt][nt][r];
        }
    }
}

// ---------------------------------------------------------------------------
// Elementwise LSTM cell update: c' = sig(f)*c + sig(i)*tanh(g); h' = sig(o)*tanh(c')
// ---------------------------------------------------------------------------
__global__ __launch_bounds__(256)
void lstm_cell_k(const float* __restrict__ gates, float* __restrict__ c,
                 __bf16* __restrict__ hbf)
{
    int idx = blockIdx.x * 256 + threadIdx.x;   // 0 .. B*H-1
    int b = idx >> 10, j = idx & (HDIM - 1);
    const float* g = gates + (size_t)b * FOURH;
    float ig = g[j], fg = g[HDIM + j], gg = g[2 * HDIM + j], og = g[3 * HDIM + j];
    float si = 1.0f / (1.0f + __expf(-ig));
    float sf = 1.0f / (1.0f + __expf(-fg));
    float so = 1.0f / (1.0f + __expf(-og));
    float tg = tanhf(gg);
    float cn = sf * c[idx] + si * tg;
    c[idx]   = cn;
    hbf[idx] = (__bf16)(so * tanhf(cn));
}

// ---------------------------------------------------------------------------
// 2-class head: logits = c @ W_out^T + b_out, log_softmax per row.
// One block (128 thr) per batch row. On last step also emit raw logits.
// ---------------------------------------------------------------------------
__global__ __launch_bounds__(128)
void classify_k(const float* __restrict__ c, const float* __restrict__ Wout,
                const float* __restrict__ bout, float* __restrict__ out,
                float* __restrict__ outFinal, int t)
{
    __shared__ float red0[128];
    __shared__ float red1[128];
    int b = blockIdx.x;
    const float* cr = c + (size_t)b * HDIM;
    float s0 = 0.f, s1 = 0.f;
    for (int j = threadIdx.x; j < HDIM; j += 128) {
        float cv = cr[j];
        s0 += cv * Wout[j];
        s1 += cv * Wout[HDIM + j];
    }
    red0[threadIdx.x] = s0;
    red1[threadIdx.x] = s1;
    __syncthreads();
    for (int off = 64; off > 0; off >>= 1) {
        if (threadIdx.x < off) {
            red0[threadIdx.x] += red0[threadIdx.x + off];
            red1[threadIdx.x] += red1[threadIdx.x + off];
        }
        __syncthreads();
    }
    if (threadIdx.x == 0) {
        float l0 = red0[0] + bout[0], l1 = red1[0] + bout[1];
        float m   = fmaxf(l0, l1);
        float lse = m + __logf(__expf(l0 - m) + __expf(l1 - m));
        size_t o = ((size_t)t * BSZ + b) * 2;
        out[o + 0] = l0 - lse;
        out[o + 1] = l1 - lse;
        if (outFinal) { outFinal[b * 2 + 0] = l0; outFinal[b * 2 + 1] = l1; }
    }
}

// ---------------------------------------------------------------------------
// Host launcher
// ---------------------------------------------------------------------------
extern "C" void kernel_launch(void* const* d_in, const int* in_sizes, int n_in,
                              void* d_out, int out_size, void* d_ws, size_t ws_size,
                              hipStream_t stream) {
    const int*   x    = (const int*)  d_in[0];   // [B, T]
    const float* emb  = (const float*)d_in[1];   // [VOCAB, E]
    const float* Wih  = (const float*)d_in[2];   // [4H, E]
    const float* Whh  = (const float*)d_in[3];   // [4H, H]
    const float* bih  = (const float*)d_in[4];   // [4H]
    const float* bhh  = (const float*)d_in[5];   // [4H]
    const float* Wout = (const float*)d_in[6];   // [2, H]
    const float* bout = (const float*)d_in[7];   // [2]
    const float* h0   = (const float*)d_in[8];   // [B, H]
    const float* c0   = (const float*)d_in[9];   // [B, H]
    float* out = (float*)d_out;                  // [T,B,2] then [B,2]

    // workspace layout (all 16B aligned)
    char* p = (char*)d_ws;
    __bf16* Wih_bf  = (__bf16*)p;  p += (size_t)FOURH * EDIM * 2;   // 8 MB
    __bf16* Whh_bf  = (__bf16*)p;  p += (size_t)FOURH * HDIM * 2;   // 8 MB
    __bf16* h_bf    = (__bf16*)p;  p += (size_t)BSZ   * HDIM * 2;   // 2 MB
    float*  bias    = (float*)p;   p += (size_t)FOURH * 4;          // 16 KB
    float*  c       = (float*)p;   p += (size_t)BSZ   * HDIM * 4;   // 4 MB
    float*  gates   = (float*)p;   p += (size_t)BSZ   * FOURH * 4;  // 16 MB

    // ---- one-time prep (deterministic every call)
    f32_to_bf16_k<<<(FOURH * EDIM + 255) / 256, 256, 0, stream>>>(Wih, Wih_bf, FOURH * EDIM);
    f32_to_bf16_k<<<(FOURH * HDIM + 255) / 256, 256, 0, stream>>>(Whh, Whh_bf, FOURH * HDIM);
    f32_to_bf16_k<<<(BSZ * HDIM + 255) / 256, 256, 0, stream>>>(h0, h_bf, BSZ * HDIM);
    bias_sum_k<<<(FOURH + 255) / 256, 256, 0, stream>>>(bih, bhh, bias, FOURH);
    hipMemcpyAsync(c, c0, (size_t)BSZ * HDIM * sizeof(float),
                   hipMemcpyDeviceToDevice, stream);

    // ---- recurrent chain
    dim3 ggrid(FOURH / 128, BSZ / 128);          // (32, 8)
    for (int t = 0; t < TSTEPS; ++t) {
        gemm_gates_k<<<ggrid, 256, 0, stream>>>(x, emb, Wih_bf, Whh_bf, bias,
                                                h_bf, gates, t);
        lstm_cell_k<<<(BSZ * HDIM) / 256, 256, 0, stream>>>(gates, c, h_bf);
        float* fin = (t == TSTEPS - 1) ? (out + (size_t)TSTEPS * BSZ * 2) : nullptr;
        classify_k<<<BSZ, 128, 0, stream>>>(c, Wout, bout, out, fin, t);
    }
}
